// Content_Relation_Module_33457795236594
// MI455X (gfx1250) — matrix operand order repaired and checked by reading, not compiled, not run on hardware
//
#include <hip/hip_runtime.h>
#include <stdint.h>

typedef __attribute__((ext_vector_type(16))) __bf16 v16bf;
typedef __attribute__((ext_vector_type(8)))  float  v8f;

#define CIN   256
#define C1    64
#define C2    128
#define CM    192
#define Hh    56
#define Ww    56
#define HW    3136
#define BATCH 4
#define EPSF  1e-8f

// ---------------------------------------------------------------------------
// Stage 0: precision / layout conversion.
//   Wc[m][k]    = bf16(Wcat[m][k])            (192 x 256, row-major)
//   XT[b][p][k] = bf16(x[b][k][p])            (pixel-major: GEMM B-tiles are
//                                              contiguous 32KB chunks, ready
//                                              for async-to-LDS copy)
// ---------------------------------------------------------------------------
#define XT_ELEMS ((size_t)BATCH * HW * CIN)
#define WC_ELEMS ((size_t)CM * CIN)

__global__ __launch_bounds__(256)
void convert_bf16_kernel(const float* __restrict__ x,
                         const float* __restrict__ W1,
                         const float* __restrict__ W2,
                         __bf16* __restrict__ XT,
                         __bf16* __restrict__ Wc)
{
    size_t i = (size_t)blockIdx.x * 256 + threadIdx.x;
    if (i < XT_ELEMS) {
        int k    = (int)(i & 255);
        size_t r = i >> 8;
        int p    = (int)(r % HW);
        int b    = (int)(r / HW);
        XT[i] = (__bf16)x[((size_t)b * CIN + k) * HW + p];
    } else if (i < XT_ELEMS + WC_ELEMS) {
        size_t j = i - XT_ELEMS;
        int m = (int)(j >> 8);
        int k = (int)(j & 255);
        float w = (m < C1) ? W1[m * CIN + k] : W2[(m - C1) * CIN + k];
        Wc[j] = (__bf16)w;
    }
}

// ---------------------------------------------------------------------------
// Stage 1: fused GEMM + ReLU on the WMMA pipe.
//   O[b, m, p] = relu( sum_k Wc[m,k] * x[b,k,p] ),  m in [0,192)
// Block = 128 threads (4 waves); tile M=64 (blockIdx.y), N=64 (blockIdx.x),
// batch = blockIdx.z. A and B tiles are each one contiguous 32KB region in
// global bf16, pulled into LDS with global_load_async_to_lds_b128 (ASYNCcnt).
// Wave w owns M-strip [w*16, w*16+16): preloads its 8 A fragments once and
// reuses them over 4 N-subtiles -> 32 v_wmma_f32_16x16x32_bf16 per wave.
// ---------------------------------------------------------------------------
__global__ __launch_bounds__(128)
void gemm_relu_wmma_kernel(const __bf16* __restrict__ XT,
                           const __bf16* __restrict__ Wc,
                           float* __restrict__ o1,
                           float* __restrict__ o2)
{
    __shared__ __bf16 As[64][CIN];   // 32 KB  As[m][k]
    __shared__ __bf16 Bs[64][CIN];   // 32 KB  Bs[n][k]  (B transposed)

    const int tid   = threadIdx.x;
    const int tileN = blockIdx.x * 64;
    const int tileM = blockIdx.y * 64;
    const int b     = blockIdx.z;

    const __bf16* gA = Wc + (size_t)tileM * CIN;                // 32KB contig
    const __bf16* gB = XT + ((size_t)b * HW + tileN) * CIN;     // 32KB contig

    // LDS byte offsets (generic LDS pointers carry the offset in addr[31:0])
    const uint32_t aBase = (uint32_t)(uintptr_t)(&As[0][0]);
    const uint32_t bBase = (uint32_t)(uintptr_t)(&Bs[0][0]);

    // async copy: 2 x 32KB, 16B per lane per op -> 16 ops per buffer per lane
    #pragma unroll
    for (int i = 0; i < 16; ++i) {
        uint32_t off = (uint32_t)(tid + i * 128) * 16u;
        asm volatile("global_load_async_to_lds_b128 %0, %1, %2"
                     :: "v"(aBase + off), "v"(off), "s"(gA) : "memory");
        asm volatile("global_load_async_to_lds_b128 %0, %1, %2"
                     :: "v"(bBase + off), "v"(off), "s"(gB) : "memory");
    }
    asm volatile("s_wait_asynccnt 0x0" ::: "memory");
    __syncthreads();

    const int wave = tid >> 5;
    const int lane = tid & 31;
    const int half = lane >> 4;         // 0: lanes 0-15, 1: lanes 16-31
    const int r    = lane & 15;
    const int mrow = wave * 16 + r;

    // preload the wave's 8 A fragments (contiguous K order, matching B:
    // lane half 0 holds K 0..15, half 1 holds K 16..31)
    v16bf afr[8];
    #pragma unroll
    for (int kk = 0; kk < 8; ++kk) {
        #pragma unroll
        for (int j = 0; j < 16; ++j)
            afr[kk][j] = As[mrow][kk * 32 + half * 16 + j];
    }

    #pragma unroll
    for (int ni = 0; ni < 4; ++ni) {
        v8f acc = {};
        #pragma unroll
        for (int kk = 0; kk < 8; ++kk) {
            // B 32x16: lanes 0-15 hold K=0..15, lanes 16-31 K=16..31 (col = r)
            v16bf bfr;
            #pragma unroll
            for (int j = 0; j < 16; ++j)
                bfr[j] = Bs[ni * 16 + r][kk * 32 + half * 16 + j];
            acc = __builtin_amdgcn_wmma_f32_16x16x32_bf16(
                      false, afr[kk], false, bfr, (short)0, acc, false, false);
        }
        // C/D layout: VGPR v -> row (v + 8*half), col = r
        #pragma unroll
        for (int v = 0; v < 8; ++v) {
            int m = tileM + wave * 16 + v + 8 * half;
            int n = tileN + ni * 16 + r;
            float val = fmaxf(acc[v], 0.0f);
            if (m < C1) o1[((size_t)b * C1 + m)        * HW + n] = val;
            else        o2[((size_t)b * C2 + (m - C1)) * HW + n] = val;
        }
    }
}

// ---------------------------------------------------------------------------
// Stage 2+3 fused: cosine relations over the 7x7 neighborhood, then
// relation-weighted aggregation of o2. One thread per pixel; o1/o2 working
// set (9.6 MB) is L2-resident on MI455X (192 MB L2), so gathers hit L2.
// ---------------------------------------------------------------------------
__global__ __launch_bounds__(224)
void relation_agg_kernel(const float* __restrict__ o1,
                         const float* __restrict__ o2,
                         float* __restrict__ out)
{
    __shared__ float relS[49][224];   // 43.9 KB

    const int tid = threadIdx.x;
    const int b   = blockIdx.y;
    const int p   = blockIdx.x * 224 + tid;      // 14 * 224 == 3136 exactly
    const int y   = p / Ww;
    const int xx  = p - y * Ww;

    const float* o1b = o1 + (size_t)b * C1 * HW;
    const float* o2b = o2 + (size_t)b * C2 * HW;

    // center feature + its norm (cen[] fully unrolled -> stays in VGPRs)
    float cen[C1];
    float n0sq = 0.0f;
    #pragma unroll
    for (int c = 0; c < C1; ++c) {
        float v = o1b[c * HW + p];
        cen[c] = v;
        n0sq += v * v;
    }
    const float n0 = fmaxf(sqrtf(n0sq), EPSF);

    // rel[t] = <cen, o1[neigh_t]> / (n0 * max(||o1[neigh_t]||, eps))
    for (int t = 0; t < 49; ++t) {
        int dy = t / 7 - 3;
        int dx = t - (t / 7) * 7 - 3;
        int ny = y + dy, nx = xx + dx;
        bool ok = ((unsigned)ny < (unsigned)Hh) && ((unsigned)nx < (unsigned)Ww);
        int np  = ok ? (ny * Ww + nx) : p;       // safe address; masked below
        float dot = 0.0f, nn = 0.0f;
        #pragma unroll
        for (int c = 0; c < C1; ++c) {
            float v = o1b[c * HW + np];
            dot += cen[c] * v;
            nn  += v * v;
        }
        float rel = dot / (n0 * fmaxf(sqrtf(nn), EPSF));
        relS[t][tid] = ok ? rel : 0.0f;          // padded taps -> 0
    }

    // neighbor indices, fully unrolled so they live in registers
    int npA[49];
    #pragma unroll
    for (int t = 0; t < 49; ++t) {
        int dy = t / 7 - 3;
        int dx = t % 7 - 3;
        int ny = y + dy, nx = xx + dx;
        bool ok = ((unsigned)ny < (unsigned)Hh) && ((unsigned)nx < (unsigned)Ww);
        npA[t] = ok ? (ny * Ww + nx) : p;        // rel==0 masks invalid taps
    }

    // out[c,p] = sum_t rel[t] * o2[c, neigh_t]
    for (int c = 0; c < C2; ++c) {
        const float* row = o2b + c * HW;
        float acc = 0.0f;
        #pragma unroll
        for (int t = 0; t < 49; ++t)
            acc += relS[t][tid] * row[npA[t]];
        out[((size_t)b * C2 + c) * HW + p] = acc;
    }
}

// ---------------------------------------------------------------------------
extern "C" void kernel_launch(void* const* d_in, const int* in_sizes, int n_in,
                              void* d_out, int out_size, void* d_ws, size_t ws_size,
                              hipStream_t stream) {
    const float* x  = (const float*)d_in[0];   // (4,256,56,56)
    const float* W1 = (const float*)d_in[1];   // (64,256)
    const float* W2 = (const float*)d_in[2];   // (128,256)
    float* out = (float*)d_out;                // (4,128,56,56)

    // workspace layout (all 16B-aligned):
    float*  o1 = (float*)d_ws;                         // 4*64*3136  f32  (3.2MB)
    float*  o2 = o1 + (size_t)BATCH * C1 * HW;         // 4*128*3136 f32  (6.4MB)
    __bf16* XT = (__bf16*)(o2 + (size_t)BATCH * C2 * HW);  // bf16 x, pixel-major (6.4MB)
    __bf16* Wc = XT + XT_ELEMS;                        // bf16 [192][256] (96KB)

    size_t convN = XT_ELEMS + WC_ELEMS;
    convert_bf16_kernel<<<dim3((convN + 255) / 256), 256, 0, stream>>>(x, W1, W2, XT, Wc);

    dim3 g1(HW / 64, CM / 64, BATCH);   // (49, 3, 4)
    gemm_relu_wmma_kernel<<<g1, 128, 0, stream>>>(XT, Wc, o1, o2);

    dim3 g2(HW / 224, BATCH);           // (14, 4)
    relation_agg_kernel<<<g2, 224, 0, stream>>>(o1, o2, out);
}
